// GeometricAttention_687194768387
// MI455X (gfx1250) — matrix operand order repaired and checked by reading, not compiled, not run on hardware
//
#include <hip/hip_runtime.h>
#include <stdint.h>

// ---------------------------------------------------------------------------
// GeometricAttention for MI455X (gfx1250, wave32, WMMA f16 16x16x32).
//
// Math note: the reference's edge bias has shape [B,H,T,1] broadcast along the
// softmax axis; softmax is shift-invariant per row, so the whole edge MLP
// (W1,b1,W2,b2) provably cancels and is skipped.
//
// Pipeline (f16 storage, f32 accumulate):
//   1) cvt: x f32 -> f16 ; Wqkv/Wg/Wo f32 -> f16 TRANSPOSED (LDS-tiled) so
//      GEMM B-fragments are k-contiguous (2 x b128 per lane, like K/V).
//   2) wmma_gemm<1>: qkv = x@Wqkv + bqkv, scattered to Q[b,h,t,d], K[b,h,t,d],
//      Vt[b,h,d,t]
//   3) wmma_gemm<2>: gate = sigmoid(x@Wg + bg) scattered to [b,h,t,d]
//   4) flash_attn: per-wave 16-query tile, 32-key chunks, online softmax,
//      rel-pos bias gather, epilogue fused with gate -> attn16[bt,D]
//   5) wmma_gemm<0>: out = attn16@Wo + bo (f32 output)
//
// Workspace: ~61 MB of f16.
// ---------------------------------------------------------------------------

typedef __attribute__((ext_vector_type(16))) _Float16 v16h;
typedef __attribute__((ext_vector_type(8)))  float    v8f;

#define TT 2048
#define HH 16
#define HD 64
#define DD 1024
#define NB 2
#define QSZ (NB*HH*TT*HD)   // 4,194,304 elements

union HF16 { v16h v; uint32_t u[8]; _Float16 f[16]; };

__device__ inline v8f wmma32f16(v16h a, v16h b, v8f c) {
  return __builtin_amdgcn_wmma_f32_16x16x32_f16(false, a, false, b, (short)0, c,
                                                false, false);
}

// A-fragment: 16(M) x 32(K), source row-major, leading dim lda, K contiguous.
// Layout: lanes 0-15 m=lane, vgpr0..7 k={0,2,..,14}+pair (contiguous),
// lanes 16-31 add +16 to k.
__device__ inline v16h load_a_rm(const _Float16* base, int lda) {
  int lane = threadIdx.x & 31;
  int m = lane & 15, grp = lane >> 4;
  HF16 a;
  const _Float16* p = base + (size_t)m * lda + grp * 16;
#pragma unroll
  for (int vg = 0; vg < 8; ++vg) {
    int k = 2 * vg;
    a.u[vg] = *(const uint32_t*)(p + k);
  }
  return a.v;
}

// B-fragment: 32(K) x 16(N), K contiguous in memory (stride 1), N stride strideN.
// Lanes 0-15 hold K=0..15 for n=lane; lanes 16-31 hold K=16..31.
__device__ inline v16h load_b_kcont(const _Float16* base, int strideN) {
  int lane = threadIdx.x & 31;
  int n = lane & 15, grp = lane >> 4;
  HF16 b;
  const _Float16* p = base + (size_t)n * strideN + grp * 16;
#pragma unroll
  for (int vg = 0; vg < 8; ++vg) b.u[vg] = *(const uint32_t*)(p + 2 * vg);
  return b.v;
}

__global__ void cvt_f32_f16(const float* __restrict__ in,
                            _Float16* __restrict__ out, int n) {
  int i = blockIdx.x * blockDim.x + threadIdx.x;
  if (i < n) out[i] = (_Float16)in[i];
}

// Transpose f32 [K,N] -> f16 [N,K], 32x32 LDS tiles, coalesced both sides.
__global__ __launch_bounds__(256) void transpose_f32_f16(
    const float* __restrict__ in, _Float16* __restrict__ out, int K, int N) {
  __shared__ float tile[32][33];
  int tx = threadIdx.x & 31;
  int ty = threadIdx.x >> 5;  // 0..7
  int n0 = blockIdx.x * 32;
  int k0 = blockIdx.y * 32;
#pragma unroll
  for (int i = 0; i < 4; ++i) {
    int k = k0 + ty + i * 8;
    tile[ty + i * 8][tx] = in[(size_t)k * N + n0 + tx];
  }
  __syncthreads();
#pragma unroll
  for (int i = 0; i < 4; ++i) {
    int n = n0 + ty + i * 8;
    out[(size_t)n * K + k0 + tx] = (_Float16)tile[tx][ty + i * 8];
  }
}

// MODE 0: f32 row-major out.  MODE 1: qkv scatter (q/k row, v transposed).
// MODE 2: sigmoid + gate scatter to [b,h,t,d].
// Bt is the TRANSPOSED weight: f16 [N, K] row-major.
template <int MODE>
__global__ __launch_bounds__(256) void wmma_gemm(
    const _Float16* __restrict__ A, const _Float16* __restrict__ Bt,
    const float* __restrict__ bias, float* __restrict__ outF,
    _Float16* __restrict__ outH, int M, int N, int K) {
  int wave = (int)((blockIdx.x * blockDim.x + threadIdx.x) >> 5);
  int tilesN = N >> 6;
  int mt = wave / tilesN, nt = wave % tilesN;
  int lane = threadIdx.x & 31, nn = lane & 15, grp = lane >> 4;

  v8f acc[4];
#pragma unroll
  for (int j = 0; j < 4; ++j)
#pragma unroll
    for (int r = 0; r < 8; ++r) acc[j][r] = 0.f;

  const _Float16* Arow = A + (size_t)(mt * 16) * K;
  for (int k0 = 0; k0 < K; k0 += 32) {
    v16h af = load_a_rm(Arow + k0, K);
#pragma unroll
    for (int j = 0; j < 4; ++j) {
      v16h bf = load_b_kcont(Bt + (size_t)(nt * 64 + j * 16) * K + k0, K);
      acc[j] = wmma32f16(af, bf, acc[j]);
    }
  }

#pragma unroll
  for (int j = 0; j < 4; ++j) {
#pragma unroll
    for (int r = 0; r < 8; ++r) {
      int row = mt * 16 + r + 8 * grp;
      int col = nt * 64 + j * 16 + nn;
      float v = acc[j][r] + bias[col];
      if constexpr (MODE == 0) {
        outF[(size_t)row * N + col] = v;
      } else if constexpr (MODE == 1) {
        int part = col >> 10, d = col & 1023;
        int hh = d >> 6, e = d & 63;
        int bb = row >> 11, t = row & 2047;
        size_t bh = (size_t)(bb * HH + hh);
        _Float16 hv = (_Float16)v;
        if (part == 0)      outH[(bh * TT + t) * HD + e] = hv;               // Q
        else if (part == 1) outH[(size_t)QSZ + (bh * TT + t) * HD + e] = hv; // K
        else                outH[(size_t)2 * QSZ + (bh * HD + e) * TT + t] = hv; // V^T
      } else {
        float g = 1.f / (1.f + __expf(-v));
        int hh = col >> 6, e = col & 63;
        int bb = row >> 11, t = row & 2047;
        size_t bh = (size_t)(bb * HH + hh);
        outH[(bh * TT + t) * HD + e] = (_Float16)g;
      }
    }
  }
}

__global__ __launch_bounds__(256) void flash_attn(
    const _Float16* __restrict__ q16, const _Float16* __restrict__ k16,
    const _Float16* __restrict__ vT, const _Float16* __restrict__ gate16,
    const float* __restrict__ relpos, _Float16* __restrict__ attn16) {
  __shared__ alignas(16) _Float16 smem[8][16 * 32];  // per-wave P staging
  int wib = threadIdx.x >> 5;
  int wave = blockIdx.x * 8 + wib;
  int qt = wave & (TT / 16 - 1);  // 128 query tiles
  int bh = wave >> 7;             // 0..31
  int h = bh & (HH - 1);
  int b = bh >> 4;
  int lane = threadIdx.x & 31, nn = lane & 15, grp = lane >> 4;
  int q0 = qt * 16;

  const _Float16* Qb = q16 + ((size_t)bh * TT + q0) * HD;
  const _Float16* Kb = k16 + (size_t)bh * TT * HD;
  const _Float16* Vb = vT + (size_t)bh * HD * TT;

  v16h qa0 = load_a_rm(Qb, HD);
  v16h qa1 = load_a_rm(Qb + 32, HD);

  float mrun[8], lrun[8], srow[2][8];
  v8f o[4];
#pragma unroll
  for (int r = 0; r < 8; ++r) { mrun[r] = -1e30f; lrun[r] = 0.f; }
#pragma unroll
  for (int j = 0; j < 4; ++j)
#pragma unroll
    for (int r = 0; r < 8; ++r) o[j][r] = 0.f;

  const float scale = 0.125f;  // 1/sqrt(64)
  _Float16* pl = smem[wib];

  for (int kt = 0; kt < TT; kt += 32) {
    // S = Q K^T for 32 keys (two 16-key subtiles), + rel-pos bias
#pragma unroll
    for (int sub = 0; sub < 2; ++sub) {
      int kb = kt + sub * 16;
      v16h kf0 = load_b_kcont(Kb + (size_t)kb * HD, HD);       // d = 0..31
      v16h kf1 = load_b_kcont(Kb + (size_t)kb * HD + 32, HD);  // d = 32..63
      v8f s;
#pragma unroll
      for (int r = 0; r < 8; ++r) s[r] = 0.f;
      s = wmma32f16(qa0, kf0, s);
      s = wmma32f16(qa1, kf1, s);
      int kk = kb + nn;
#pragma unroll
      for (int r = 0; r < 8; ++r) {
        int qq = q0 + r + 8 * grp;
        int rel = qq - kk;
        rel = rel < -255 ? -255 : (rel > 255 ? 255 : rel);
        srow[sub][r] = s[r] * scale + relpos[(rel + 255) * HH + h];
      }
    }
    // Online softmax over the 32 new columns (row = (r, grp); 16-lane reduce)
    float alpha[8];
#pragma unroll
    for (int r = 0; r < 8; ++r) {
      float mx = fmaxf(srow[0][r], srow[1][r]);
#pragma unroll
      for (int off = 1; off < 16; off <<= 1) mx = fmaxf(mx, __shfl_xor(mx, off, 32));
      float mn = fmaxf(mrun[r], mx);
      float a = __expf(mrun[r] - mn);
      float p0 = __expf(srow[0][r] - mn);
      float p1 = __expf(srow[1][r] - mn);
      float rs = p0 + p1;
#pragma unroll
      for (int off = 1; off < 16; off <<= 1) rs += __shfl_xor(rs, off, 32);
      mrun[r] = mn;
      lrun[r] = lrun[r] * a + rs;
      alpha[r] = a;
      srow[0][r] = p0;
      srow[1][r] = p1;
    }
#pragma unroll
    for (int j = 0; j < 4; ++j)
#pragma unroll
      for (int r = 0; r < 8; ++r) o[j][r] *= alpha[r];

    // Stage P (16x32) through LDS to re-layout C-frag -> A-frag
#pragma unroll
    for (int sub = 0; sub < 2; ++sub)
#pragma unroll
      for (int r = 0; r < 8; ++r)
        pl[(r + 8 * grp) * 32 + sub * 16 + nn] = (_Float16)srow[sub][r];
    asm volatile("s_wait_dscnt 0" ::: "memory");
    v16h pf = load_a_rm(pl, 32);

    // O += P @ V  (V^T layout: key index contiguous -> dword B-frag loads)
#pragma unroll
    for (int j = 0; j < 4; ++j) {
      v16h vf = load_b_kcont(Vb + (size_t)(j * 16) * TT + kt, TT);
      o[j] = wmma32f16(pf, vf, o[j]);
    }
  }

  // Epilogue: normalize, gate, store f16 [bt, D]
  const _Float16* Gb = gate16 + ((size_t)bh * TT + q0) * HD;
#pragma unroll
  for (int j = 0; j < 4; ++j) {
#pragma unroll
    for (int r = 0; r < 8; ++r) {
      int mq = r + 8 * grp;
      float val = o[j][r] / lrun[r];
      float g = (float)Gb[(size_t)mq * HD + j * 16 + nn];
      int row = b * TT + q0 + mq;
      attn16[(size_t)row * DD + h * HD + j * 16 + nn] = (_Float16)(val * g);
    }
  }
}

extern "C" void kernel_launch(void* const* d_in, const int* in_sizes, int n_in,
                              void* d_out, int out_size, void* d_ws, size_t ws_size,
                              hipStream_t stream) {
  const float* x      = (const float*)d_in[0];
  const float* Wqkv   = (const float*)d_in[1];
  const float* bqkv   = (const float*)d_in[2];
  const float* Wg     = (const float*)d_in[3];
  const float* bg     = (const float*)d_in[4];
  const float* relpos = (const float*)d_in[5];
  // d_in[6..9] = W1, b1, W2, b2: unused — edge bias is constant along the
  // softmax axis, hence a provable no-op.
  const float* Wo = (const float*)d_in[10];
  const float* bo = (const float*)d_in[11];
  float* out = (float*)d_out;

  _Float16* ws = (_Float16*)d_ws;
  size_t off = 0;
  _Float16* x16    = ws + off; off += (size_t)4096 * 1024;
  _Float16* WqT    = ws + off; off += (size_t)1024 * 3072;  // [3072,1024]
  _Float16* WgT    = ws + off; off += (size_t)1024 * 1024;  // [1024,1024]
  _Float16* WoT    = ws + off; off += (size_t)1024 * 1024;  // [1024,1024]
  _Float16* q16    = ws + off; off += (size_t)3 * QSZ;      // Q, K, V^T
  _Float16* gate16 = ws + off; off += (size_t)QSZ;
  _Float16* attn16 = ws + off; off += (size_t)QSZ;          // total ~61 MB

  cvt_f32_f16<<<(4096 * 1024) / 256, 256, 0, stream>>>(x, x16, 4096 * 1024);
  transpose_f32_f16<<<dim3(3072 / 32, 1024 / 32), 256, 0, stream>>>(Wqkv, WqT, 1024, 3072);
  transpose_f32_f16<<<dim3(1024 / 32, 1024 / 32), 256, 0, stream>>>(Wg, WgT, 1024, 1024);
  transpose_f32_f16<<<dim3(1024 / 32, 1024 / 32), 256, 0, stream>>>(Wo, WoT, 1024, 1024);

  {  // qkv projection + head scatter
    int M = 4096, N = 3072, K = 1024;
    int waves = (M / 16) * (N / 64);
    wmma_gemm<1><<<waves / 8, 256, 0, stream>>>(x16, WqT, bqkv, nullptr, q16, M, N, K);
  }
  {  // gate = sigmoid(x@Wg + bg)
    int M = 4096, N = 1024, K = 1024;
    int waves = (M / 16) * (N / 64);
    wmma_gemm<2><<<waves / 8, 256, 0, stream>>>(x16, WgT, bg, nullptr, gate16, M, N, K);
  }
  {  // flash attention, gate fused into epilogue
    int waves = NB * HH * (TT / 16);  // 4096
    flash_attn<<<waves / 8, 256, 0, stream>>>(q16, q16 + QSZ, q16 + (size_t)2 * QSZ,
                                              gate16, relpos, attn16);
  }
  {  // output projection (f32 out)
    int M = 4096, N = 1024, K = 1024;
    int waves = (M / 16) * (N / 64);
    wmma_gemm<0><<<waves / 8, 256, 0, stream>>>(attn16, WoT, bo, out, nullptr, M, N, K);
  }
  (void)in_sizes; (void)n_in; (void)out_size; (void)ws_size;
}
